// LinkedSeq2Seq_18614388261036
// MI455X (gfx1250) — matrix operand order repaired and checked by reading, not compiled, not run on hardware
//
#include <hip/hip_runtime.h>
#include <hip/hip_bf16.h>

static constexpr int kH = 2048;   // hidden
static constexpr int kE = 2048;   // embed
static constexpr int kV = 512;    // productions
static constexpr int kT = 4096;   // encoder tokens
static constexpr int kS = 20;     // decode steps

typedef __attribute__((ext_vector_type(16))) __bf16 v16bf;
typedef __attribute__((ext_vector_type(8)))  float  v8f;
typedef unsigned int u32x4 __attribute__((ext_vector_type(4)));

union V16 { v16bf v; u32x4 q[2]; };

__device__ __forceinline__ unsigned short f2bf(float f) {
  unsigned int u = __float_as_uint(f);
  u += 0x7FFFu + ((u >> 16) & 1u);          // round-to-nearest-even
  return (unsigned short)(u >> 16);
}
__device__ __forceinline__ float bf2f(unsigned short h) {
  return __uint_as_float(((unsigned int)h) << 16);
}

// ---------------------------------------------------------------- converts
__global__ void k_f32_to_bf16(const float* __restrict__ src,
                              unsigned short* __restrict__ dst, int n) {
  for (int i = blockIdx.x * blockDim.x + threadIdx.x; i < n;
       i += gridDim.x * blockDim.x)
    dst[i] = f2bf(src[i]);
}

__global__ void k_init(const float* __restrict__ token_rep,
                       const float* __restrict__ first_action,
                       unsigned short* __restrict__ feat_bf,   // [2H] (h | att_ret)
                       unsigned short* __restrict__ a_bf,      // [E]
                       float* __restrict__ c) {                // [H]
  int i = blockIdx.x * blockDim.x + threadIdx.x;
  if (i < kH) { feat_bf[i] = f2bf(token_rep[i]); c[i] = 0.f; }
  if (i < kE) a_bf[i] = f2bf(first_action[i]);
}

// ---------------------------------------------------------------- WMMA GEMV
// y[M] = W[M,K](bf16,row-major) * x[K](bf16)  [+ bias], epilogue modes:
//   mode 0: store, mode 1: y += result, mode 2: y = tanh(result) (+ bf16 copy)
//
// Block = 8 waves = 2 row-tiles (16 rows each) x 4 K-quarters; per-wave partial
// sums are combined through LDS. x is staged once per block into LDS via
// GLOBAL_LOAD_ASYNC_TO_LDS_B128 (ASYNCcnt) and the replicated B operand is fed
// from LDS. Inner loop: 4 independent v_wmma_f32_16x16x32_bf16 per iteration.
__global__ void __launch_bounds__(256)
k_gemv_wmma(const unsigned short* __restrict__ W,
            const unsigned short* __restrict__ x,
            const float* __restrict__ bias,
            float* __restrict__ y,
            unsigned short* __restrict__ ybf,
            int M, int K, int mode) {
  __shared__ __align__(16) unsigned short sx[4096];  // staged x (up to 8KB)
  __shared__ float sred[8][16];                      // per-wave partial rows

  const int lane = threadIdx.x & 31;
  const int wave = threadIdx.x >> 5;      // 8 waves
  const int kq   = wave >> 1;             // K quarter 0..3
  const int rt   = wave & 1;              // row tile 0..1
  const int row0 = (blockIdx.x * 2 + rt) * 16;
  const int half = lane >> 4;             // 0: lanes 0-15, 1: lanes 16-31
  const int r    = lane & 15;             // A-matrix row within tile

  // ---- stage x[0:K] into LDS with async tensor-path copies (wave 0 only)
  if (wave == 0) {
    unsigned ldsBase = (unsigned)(size_t)(void*)sx;   // flat low 32 = LDS addr
    const int nIssue = K >> 8;                        // K*2 bytes / 512 per issue
    for (int i = 0; i < nIssue; ++i) {
      unsigned off  = (unsigned)((i * 32 + lane) * 16);
      unsigned dsad = ldsBase + off;
      asm volatile("global_load_async_to_lds_b128 %0, %1, %2"
                   :: "v"(dsad), "v"(off), "s"(x) : "memory");
    }
    asm volatile("s_wait_asynccnt 0" ::: "memory");
  }
  __syncthreads();

  const unsigned short* wrow = W + (size_t)(row0 + r) * (size_t)K;
  const int kbeg = kq * (K >> 2);
  const int kend = kbeg + (K >> 2);

  v8f acc[4] = {};
  for (int kc = kbeg; kc < kend; kc += 128) {
#pragma unroll
    for (int u = 0; u < 4; ++u) {
      const int k = kc + u * 32;
      V16 a, b;
      // A 16x32 bf16 tile: half=0 lane holds K = k+0..7 and k+16..23
      a.q[0] = *(const u32x4*)(wrow + k      + half * 8);
      a.q[1] = *(const u32x4*)(wrow + k + 16 + half * 8);
      // B 32x16 bf16, x replicated in every column (from LDS broadcast)
      b.q[0] = *(const u32x4*)(sx + k      + half * 8);
      b.q[1] = *(const u32x4*)(sx + k + 16 + half * 8);
      acc[u] = __builtin_amdgcn_wmma_f32_16x16x32_bf16(false, a.v, false, b.v,
                                                       (short)0, acc[u], false,
                                                       false);
    }
    __builtin_prefetch(wrow + kc + 512, 0, 1);   // global_prefetch_b8
  }

  // C layout: VGPR j, lanes 0-15 -> row j; lanes 16-31 -> row 8+j
  if (r == 0) {
#pragma unroll
    for (int j = 0; j < 8; ++j)
      sred[wave][half * 8 + j] = (acc[0][j] + acc[1][j]) + (acc[2][j] + acc[3][j]);
  }
  __syncthreads();

  const int t = threadIdx.x;
  if (t < 32) {
    const int rtt = t >> 4, i = t & 15;
    const int row = (blockIdx.x * 2 + rtt) * 16 + i;
    float v = (sred[rtt][i] + sred[rtt + 2][i]) +
              (sred[rtt + 4][i] + sred[rtt + 6][i]);
    if (bias) v += bias[row];
    if (mode == 1) v += y[row];
    if (mode == 2) v = tanhf(v);
    if (y)   y[row]   = v;
    if (ybf) ybf[row] = f2bf(v);
  }
}

// ---------------------------------------------------------------- LSTM cell
__global__ void k_lstm(const float* __restrict__ gates,   // [4H] i,f,g,o
                       float* __restrict__ c,             // [H] in/out
                       unsigned short* __restrict__ feat_bf) { // h2 -> feat[0:H]
  int j = blockIdx.x * blockDim.x + threadIdx.x;
  if (j >= kH) return;
  float i = gates[j], f = gates[kH + j], g = gates[2 * kH + j], o = gates[3 * kH + j];
  i = 1.f / (1.f + expf(-i));
  f = 1.f / (1.f + expf(-f));
  o = 1.f / (1.f + expf(-o));
  g = tanhf(g);
  float c2 = f * c[j] + i * g;
  c[j] = c2;
  feat_bf[j] = f2bf(o * tanhf(c2));
}

// ------------------------------------------------------- softmax over T=4096
__global__ void k_softmax_T(const float* __restrict__ logits,
                            float* __restrict__ w) {
  __shared__ float sm[1024];
  const int tid = threadIdx.x;
  float m = -3.4e38f;
  for (int t = tid; t < kT; t += 1024) m = fmaxf(m, logits[t]);
  sm[tid] = m; __syncthreads();
  for (int s = 512; s > 0; s >>= 1) {
    if (tid < s) sm[tid] = fmaxf(sm[tid], sm[tid + s]);
    __syncthreads();
  }
  m = sm[0]; __syncthreads();
  float z = 0.f;
  for (int t = tid; t < kT; t += 1024) z += expf(logits[t] - m);
  sm[tid] = z; __syncthreads();
  for (int s = 512; s > 0; s >>= 1) {
    if (tid < s) sm[tid] += sm[tid + s];
    __syncthreads();
  }
  const float inv = 1.f / sm[0];
  for (int t = tid; t < kT; t += 1024) w[t] = expf(logits[t] - m) * inv;
}

// ------------------------------------------- att_ret[j] = sum_t w[t]*E[t,j]
__global__ void k_att_ret(const float* __restrict__ w,
                          const unsigned short* __restrict__ Ebf, // [T,H] bf16
                          unsigned short* __restrict__ feat_bf) { // -> feat[H:2H]
  __shared__ float sm[4][64];
  const int c64 = threadIdx.x & 63;
  const int tg  = threadIdx.x >> 6;             // 0..3 t-slices
  const int col = blockIdx.x * 64 + c64;
  float acc = 0.f;
  for (int t = tg; t < kT; t += 4)
    acc += w[t] * bf2f(Ebf[(size_t)t * kH + col]);
  sm[tg][c64] = acc;
  __syncthreads();
  if (tg == 0)
    feat_bf[kH + col] = f2bf(sm[0][c64] + sm[1][c64] + sm[2][c64] + sm[3][c64]);
}

// --------------------------- masked log-softmax, ll accumulate, next action
__global__ void k_finalize(const float* __restrict__ score,        // [V]
                           const unsigned char* __restrict__ mask, // [V]
                           const int* __restrict__ gold_ids, int step,
                           const unsigned short* __restrict__ embed_bf, // [V,E]
                           unsigned short* __restrict__ a_bf,      // [E]
                           float* __restrict__ out) {
  __shared__ float sm[kV];
  __shared__ float sGold;
  const int j = threadIdx.x;                    // 512 threads
  const int gold = gold_ids[step];
  const float masked = mask[j] ? score[j] : -1e30f;
  if (j == gold) sGold = masked;
  sm[j] = masked; __syncthreads();
  for (int s = kV / 2; s > 0; s >>= 1) {
    if (j < s) sm[j] = fmaxf(sm[j], sm[j + s]);
    __syncthreads();
  }
  const float m = sm[0]; __syncthreads();
  sm[j] = expf(masked - m); __syncthreads();
  for (int s = kV / 2; s > 0; s >>= 1) {
    if (j < s) sm[j] += sm[j + s];
    __syncthreads();
  }
  if (j == 0) {
    const float ll = sGold - m - logf(sm[0]);
    out[0] = (step == 0) ? ll : (out[0] + ll);
  }
  const unsigned short* src = embed_bf + (size_t)gold * kE;
  for (int i = j; i < kE; i += kV) a_bf[i] = src[i];
}

// ----------------------------------------------------------------- launcher
extern "C" void kernel_launch(void* const* d_in, const int* in_sizes, int n_in,
                              void* d_out, int out_size, void* d_ws, size_t ws_size,
                              hipStream_t stream) {
  const float* token_rep     = (const float*)d_in[0];
  const float* token_encodes = (const float*)d_in[1];
  const int*   gold_ids      = (const int*)d_in[2];
  const unsigned char* cand_mask = (const unsigned char*)d_in[3];
  const float* W_ih   = (const float*)d_in[4];
  const float* W_hh   = (const float*)d_in[5];
  const float* b_ih   = (const float*)d_in[6];
  const float* b_hh   = (const float*)d_in[7];
  const float* mlp1_W = (const float*)d_in[8];
  const float* mlp1_b = (const float*)d_in[9];
  const float* mlp2_W = (const float*)d_in[10];
  const float* mlp2_b = (const float*)d_in[11];
  const float* embed  = (const float*)d_in[12];
  const float* first_action = (const float*)d_in[13];
  float* out = (float*)d_out;

  // workspace carve-up (256B aligned slabs)
  char* ws = (char*)d_ws;
  auto aUS = [&](size_t n) { unsigned short* p = (unsigned short*)ws;
                             ws += ((n * 2 + 255) / 256) * 256; return p; };
  auto aF  = [&](size_t n) { float* p = (float*)ws;
                             ws += ((n * 4 + 255) / 256) * 256; return p; };

  unsigned short* wih_bf   = aUS((size_t)4 * kH * kE);
  unsigned short* whh_bf   = aUS((size_t)4 * kH * kH);
  unsigned short* tokE_bf  = aUS((size_t)kT * kH);
  unsigned short* mlp1_bf  = aUS((size_t)kH * 2 * kH);
  unsigned short* mlp2_bf  = aUS((size_t)kV * kH);
  unsigned short* embed_bf = aUS((size_t)kV * kE);
  unsigned short* feat_bf  = aUS(2 * kH);   // [h2 | att_ret] bf16
  unsigned short* a_bf     = aUS(kE);
  unsigned short* hid_bf   = aUS(kH);
  float* cbuf    = aF(kH);
  float* gates   = aF(4 * kH);
  float* att_lg  = aF(kT);
  float* attw    = aF(kT);
  float* hid_f   = aF(kH);
  float* score   = aF(kV);

  // one-time (per launch) fp32 -> bf16 weight conversion; working set (~100MB)
  // then lives in the 192MB L2 for all 20 steps.
  k_f32_to_bf16<<<512, 256, 0, stream>>>(W_ih,   wih_bf,   4 * kH * kE);
  k_f32_to_bf16<<<512, 256, 0, stream>>>(W_hh,   whh_bf,   4 * kH * kH);
  k_f32_to_bf16<<<512, 256, 0, stream>>>(token_encodes, tokE_bf, kT * kH);
  k_f32_to_bf16<<<512, 256, 0, stream>>>(mlp1_W, mlp1_bf,  kH * 2 * kH);
  k_f32_to_bf16<<<512, 256, 0, stream>>>(mlp2_W, mlp2_bf,  kV * kH);
  k_f32_to_bf16<<<512, 256, 0, stream>>>(embed,  embed_bf, kV * kE);
  k_init<<<8, 256, 0, stream>>>(token_rep, first_action, feat_bf, a_bf, cbuf);

  for (int s = 0; s < kS; ++s) {
    // gates = W_ih@a + b_ih + W_hh@h + b_hh   (M=8192)
    k_gemv_wmma<<<256, 256, 0, stream>>>(wih_bf, a_bf,   b_ih, gates, nullptr,
                                         4 * kH, kE, 0);
    k_gemv_wmma<<<256, 256, 0, stream>>>(whh_bf, feat_bf, b_hh, gates, nullptr,
                                         4 * kH, kH, 1);
    k_lstm<<<8, 256, 0, stream>>>(gates, cbuf, feat_bf);
    // attention logits = token_encodes @ h2   (M=4096)
    k_gemv_wmma<<<128, 256, 0, stream>>>(tokE_bf, feat_bf, nullptr, att_lg,
                                         nullptr, kT, kH, 0);
    k_softmax_T<<<1, 1024, 0, stream>>>(att_lg, attw);
    k_att_ret<<<32, 256, 0, stream>>>(attw, tokE_bf, feat_bf);
    // hidden = tanh(mlp1 @ feat + b1)         (M=2048, K=4096)
    k_gemv_wmma<<<64, 256, 0, stream>>>(mlp1_bf, feat_bf, mlp1_b, hid_f, hid_bf,
                                        kH, 2 * kH, 2);
    // score = mlp2 @ hidden + b2              (M=512)
    k_gemv_wmma<<<16, 256, 0, stream>>>(mlp2_bf, hid_bf, mlp2_b, score, nullptr,
                                        kV, kH, 0);
    k_finalize<<<1, kV, 0, stream>>>(score, cand_mask + (size_t)s * kV,
                                     gold_ids, s, embed_bf, a_bf, out);
  }
}
